// HGNNPConv_66211215835666
// MI455X (gfx1250) — compile-verified, matrix operand order checked by reading, not run on hardware
//
#include <hip/hip_runtime.h>
#include <hip/hip_bf16.h>
#include <math.h>

#define NV 20000
#define NE 4000
#define NNZ_N 160000
#define C 512

typedef __attribute__((ext_vector_type(16))) __bf16 v16bf;
typedef __attribute__((ext_vector_type(8)))  __bf16 v8bf;
typedef __attribute__((ext_vector_type(8)))  float  v8f;

union Frag { v16bf v; v8bf h[2]; };

static __device__ __forceinline__ unsigned short f2bf_rne(float f) {
  unsigned int u = __float_as_uint(f);
  u += 0x7FFFu + ((u >> 16) & 1u);   // round-to-nearest-even
  return (unsigned short)(u >> 16);
}

// ---------------- zero / convert / relu ----------------

__global__ void hg_zero_kernel(float* __restrict__ p, int n) {
  int i = blockIdx.x * blockDim.x + threadIdx.x;
  if (i < n) p[i] = 0.0f;
}

__global__ void hg_f32_to_bf16_kernel(const float* __restrict__ src,
                                      unsigned short* __restrict__ dst, int n) {
  int i = blockIdx.x * blockDim.x + threadIdx.x;
  if (i < n) dst[i] = f2bf_rne(src[i]);
}

__global__ void hg_relu_kernel(float* __restrict__ p, int n) {
  int i = blockIdx.x * blockDim.x + threadIdx.x;
  if (i < n) p[i] = fmaxf(p[i], 0.0f);
}

// ---------------- segment stats ----------------

// pass 1: edge degree + per-vertex max of e2v_weight (weights >= 0 -> uint-bit max)
__global__ void hg_stats1_kernel(const int* __restrict__ v_idx,
                                 const int* __restrict__ e_idx,
                                 const float* __restrict__ w,
                                 int* __restrict__ deg,
                                 unsigned int* __restrict__ vmax_bits) {
  int k = blockIdx.x * blockDim.x + threadIdx.x;
  if (k >= NNZ_N) return;
  atomicAdd(&deg[e_idx[k]], 1);
  atomicMax(&vmax_bits[v_idx[k]], __float_as_uint(w[k]));
}

// pass 2: per-vertex sum of exp(w - max)
__global__ void hg_stats2_kernel(const int* __restrict__ v_idx,
                                 const float* __restrict__ w,
                                 const unsigned int* __restrict__ vmax_bits,
                                 float* __restrict__ vsum) {
  int k = blockIdx.x * blockDim.x + threadIdx.x;
  if (k >= NNZ_N) return;
  int v = v_idx[k];
  float m = __uint_as_float(vmax_bits[v]);
  atomicAdd(&vsum[v], __expf(w[k] - m));
}

// ---------------- bf16 WMMA GEMM: Xp = X * W^T + b ----------------
// One wave -> 16x64 output strip. Register double-buffered K loop: fragments
// for K-step i+1 are in flight while the 4 WMMAs of step i execute, so the
// s_wait before each WMMA group is partial instead of loadcnt==0.
__global__ void __launch_bounds__(256)
hg_gemm_wmma_kernel(const unsigned short* __restrict__ Xh,   // [NV, C] bf16 bits
                    const unsigned short* __restrict__ Wh,   // [C, C] bf16 bits, row = out ch
                    const float* __restrict__ bias,
                    float* __restrict__ Xp)                  // [NV, C] f32
{
  const int wave = threadIdx.x >> 5;        // 0..7 -> 64-col group
  const int lane = threadIdx.x & 31;
  const int half = lane >> 4;               // K-half selector per ISA layout
  const int l    = lane & 15;
  const int tileM = blockIdx.x * 16;
  const int tileN = wave * 64;

  v8f acc0 = {}, acc1 = {}, acc2 = {}, acc3 = {};

  const unsigned short* aRow  = Xh + (size_t)(tileM + l) * C;
  const unsigned short* bRow0 = Wh + (size_t)(tileN + l) * C;
  const unsigned short* bRow1 = bRow0 + (size_t)16 * C;
  const unsigned short* bRow2 = bRow0 + (size_t)32 * C;
  const unsigned short* bRow3 = bRow0 + (size_t)48 * C;

  Frag aA, b0A, b1A, b2A, b3A;   // buffer A (even K-steps)
  Frag aB, b0B, b1B, b2B, b3B;   // buffer B (odd  K-steps)

#define LOAD_FRAGS(FA, F0, F1, F2, F3, KK)                                     \
  do {                                                                         \
    FA.h[0] = *(const v8bf*)(aRow  + (KK));  FA.h[1] = *(const v8bf*)(aRow  + (KK) + 16); \
    F0.h[0] = *(const v8bf*)(bRow0 + (KK));  F0.h[1] = *(const v8bf*)(bRow0 + (KK) + 16); \
    F1.h[0] = *(const v8bf*)(bRow1 + (KK));  F1.h[1] = *(const v8bf*)(bRow1 + (KK) + 16); \
    F2.h[0] = *(const v8bf*)(bRow2 + (KK));  F2.h[1] = *(const v8bf*)(bRow2 + (KK) + 16); \
    F3.h[0] = *(const v8bf*)(bRow3 + (KK));  F3.h[1] = *(const v8bf*)(bRow3 + (KK) + 16); \
  } while (0)

#define DO_WMMA(FA, F0, F1, F2, F3)                                            \
  do {                                                                         \
    acc0 = __builtin_amdgcn_wmma_f32_16x16x32_bf16(false, FA.v, false, F0.v, (short)0, acc0, false, false); \
    acc1 = __builtin_amdgcn_wmma_f32_16x16x32_bf16(false, FA.v, false, F1.v, (short)0, acc1, false, false); \
    acc2 = __builtin_amdgcn_wmma_f32_16x16x32_bf16(false, FA.v, false, F2.v, (short)0, acc2, false, false); \
    acc3 = __builtin_amdgcn_wmma_f32_16x16x32_bf16(false, FA.v, false, F3.v, (short)0, acc3, false, false); \
  } while (0)

  // prime buffer A with K-step 0
  LOAD_FRAGS(aA, b0A, b1A, b2A, b3A, half * 8);

  // 8 iterations, 2 K-steps (64 K) each
  #pragma unroll
  for (int k0 = 0; k0 < C; k0 += 64) {
    const int kkB = k0 + 32 + half * 8;          // always < C (last: 480+8..)
    LOAD_FRAGS(aB, b0B, b1B, b2B, b3B, kkB);     // prefetch odd step
    DO_WMMA(aA, b0A, b1A, b2A, b3A);             // consume even step
    if (k0 + 64 < C) {
      const int kkA = k0 + 64 + half * 8;
      LOAD_FRAGS(aA, b0A, b1A, b2A, b3A, kkA);   // prefetch next even step
    }
    DO_WMMA(aB, b0B, b1B, b2B, b3B);             // consume odd step
  }

#undef LOAD_FRAGS
#undef DO_WMMA

  // C/D layout: VGPR i -> row tileM+i (lanes 0-15) / tileM+8+i (lanes 16-31)
  const int col = tileN + l;
  #pragma unroll
  for (int i = 0; i < 8; ++i) {
    int row = tileM + i + half * 8;
    float* out = Xp + (size_t)row * C;
    out[col +  0] = acc0[i] + bias[col +  0];
    out[col + 16] = acc1[i] + bias[col + 16];
    out[col + 32] = acc2[i] + bias[col + 32];
    out[col + 48] = acc3[i] + bias[col + 48];
  }
}

// ---------------- scatter passes ----------------
// One block (128 threads) per nnz entry; each thread moves one float4
// (global_load_b128) and issues 4 HW f32 atomics.

__global__ void __launch_bounds__(128)
hg_v2e_kernel(const int* __restrict__ v_idx, const int* __restrict__ e_idx,
              const int* __restrict__ deg,
              const float* __restrict__ Xp, float* __restrict__ Xe) {
  const int k = blockIdx.x;
  const int v = v_idx[k], e = e_idx[k];
  const float coef = 1.0f / (float)deg[e];
  const float4 s = ((const float4*)(Xp + (size_t)v * C))[threadIdx.x];
  float* __restrict__ dst = Xe + (size_t)e * C + threadIdx.x * 4;
  unsafeAtomicAdd(&dst[0], coef * s.x);
  unsafeAtomicAdd(&dst[1], coef * s.y);
  unsafeAtomicAdd(&dst[2], coef * s.z);
  unsafeAtomicAdd(&dst[3], coef * s.w);
}

__global__ void __launch_bounds__(128)
hg_e2v_kernel(const int* __restrict__ v_idx, const int* __restrict__ e_idx,
              const float* __restrict__ w,
              const unsigned int* __restrict__ vmax_bits,
              const float* __restrict__ vsum,
              const float* __restrict__ Xe, float* __restrict__ out) {
  const int k = blockIdx.x;
  const int v = v_idx[k], e = e_idx[k];
  const float m = __uint_as_float(vmax_bits[v]);
  float ssum = vsum[v];
  const float coef = __expf(w[k] - m) / (ssum > 0.0f ? ssum : 1.0f);
  const float4 s = ((const float4*)(Xe + (size_t)e * C))[threadIdx.x];
  float* __restrict__ dst = out + (size_t)v * C + threadIdx.x * 4;
  unsafeAtomicAdd(&dst[0], coef * s.x);
  unsafeAtomicAdd(&dst[1], coef * s.y);
  unsafeAtomicAdd(&dst[2], coef * s.z);
  unsafeAtomicAdd(&dst[3], coef * s.w);
}

// ---------------- launch ----------------

extern "C" void kernel_launch(void* const* d_in, const int* in_sizes, int n_in,
                              void* d_out, int out_size, void* d_ws, size_t ws_size,
                              hipStream_t stream) {
  const float* X    = (const float*)d_in[0];   // [NV, C]
  const float* W    = (const float*)d_in[1];   // [C, C]
  const float* bias = (const float*)d_in[2];   // [C]
  const float* e2vw = (const float*)d_in[3];   // [NNZ]
  const int*   vidx = (const int*)d_in[4];     // [NNZ]
  const int*   eidx = (const int*)d_in[5];     // [NNZ]
  float* out = (float*)d_out;                  // [NV, C]

  // workspace layout
  char* ws = (char*)d_ws;
  unsigned short* Xh   = (unsigned short*)(ws + 0);          // 20,480,000 B
  unsigned short* Wh   = (unsigned short*)(ws + 20480000);   //    524,288 B
  float*          Xp   = (float*)(ws + 21004288);            // 40,960,000 B
  float*          Xe   = (float*)(ws + 61964288);            //  8,192,000 B
  int*            deg  = (int*)  (ws + 70156288);            //     16,000 B
  unsigned int*   vmax = (unsigned int*)(ws + 70172288);     //     80,000 B
  float*          vsum = (float*)(ws + 70252288);            //     80,000 B  (stats block contiguous)

  const int nXe   = NE * C;                        // 2,048,000
  const int nStat = (16000 + 80000 + 80000) / 4;   // contiguous deg|vmax|vsum
  const int nOut  = NV * C;                        // 10,240,000
  const int nX    = NV * C;
  const int nW    = C * C;

  // zero-init accumulators + stats + output
  hg_zero_kernel<<<(nXe   + 255) / 256, 256, 0, stream>>>(Xe, nXe);
  hg_zero_kernel<<<(nStat + 255) / 256, 256, 0, stream>>>((float*)deg, nStat);
  hg_zero_kernel<<<(nOut  + 255) / 256, 256, 0, stream>>>(out, nOut);

  // bf16 copies of X and W
  hg_f32_to_bf16_kernel<<<(nX + 255) / 256, 256, 0, stream>>>(X, Xh, nX);
  hg_f32_to_bf16_kernel<<<(nW + 255) / 256, 256, 0, stream>>>(W, Wh, nW);

  // segment stats
  hg_stats1_kernel<<<(NNZ_N + 255) / 256, 256, 0, stream>>>(vidx, eidx, e2vw, deg, vmax);
  hg_stats2_kernel<<<(NNZ_N + 255) / 256, 256, 0, stream>>>(vidx, e2vw, vmax, vsum);

  // GEMM: 1250 M-tiles x (8 waves x 64 cols)
  hg_gemm_wmma_kernel<<<NV / 16, 256, 0, stream>>>(Xh, Wh, bias, Xp);

  // scatter passes
  hg_v2e_kernel<<<NNZ_N, 128, 0, stream>>>(vidx, eidx, deg, Xp, Xe);
  hg_e2v_kernel<<<NNZ_N, 128, 0, stream>>>(vidx, eidx, e2vw, vmax, vsum, Xe, out);

  // ReLU
  hg_relu_kernel<<<(nOut + 255) / 256, 256, 0, stream>>>(out, nOut);
}
